// QuantizedLinear_42013370090058
// MI455X (gfx1250) — 8-bit matrix operand order repaired and checked by reading, compiled, not run on hardware
//
#include <hip/hip_runtime.h>
#include <stdint.h>

typedef int v8i __attribute__((ext_vector_type(8)));
typedef int v4i __attribute__((ext_vector_type(4)));

#define TOKENS 8192
#define IN_F   4096
#define OUT_F  4096
#define BM     128
#define BN     128
#define BK     64
#define LDSS   80
#define KITERS (IN_F / BK)

#if __has_builtin(__builtin_amdgcn_global_load_async_to_lds_b128)
#define HAVE_ASYNC 1
#else
#define HAVE_ASYNC 0
#endif

typedef __attribute__((address_space(1))) v4i gv4i;
typedef __attribute__((address_space(3))) v4i lv4i;

__device__ __forceinline__ void cp16(const int8_t* g, int8_t* l) {
#if HAVE_ASYNC
    __builtin_amdgcn_global_load_async_to_lds_b128(
        (gv4i*)g, (lv4i*)l, 0, 0);
#else
    *reinterpret_cast<int4*>(l) = *reinterpret_cast<const int4*>(g);
#endif
}

__device__ __forceinline__ void async_wait0() {
#if HAVE_ASYNC
#if __has_builtin(__builtin_amdgcn_s_wait_asynccnt)
    __builtin_amdgcn_s_wait_asynccnt(0);
#else
    asm volatile("s_wait_asynccnt 0" ::: "memory");
#endif
#endif
}

__device__ __forceinline__ void frag_compute(const int8_t* aT, const int8_t* bT,
                                             int aBase, int bBase,
                                             v8i (&acc)[2][4])
{
    v8i aF[2], bF[4];
    #pragma unroll
    for (int sm = 0; sm < 2; ++sm) {
        const int8_t* p = aT + aBase + sm * 16 * LDSS;
        int4 q0 = *reinterpret_cast<const int4*>(p + 0);
        int4 q1 = *reinterpret_cast<const int4*>(p + 32);
        v8i a;
        a[0] = q0.x; a[1] = q0.y; a[2] = q0.z; a[3] = q0.w;
        a[4] = q1.x; a[5] = q1.y; a[6] = q1.z; a[7] = q1.w;
        aF[sm] = a;
    }
    #pragma unroll
    for (int sn = 0; sn < 4; ++sn) {
        const int8_t* p = bT + bBase + sn * 16 * LDSS;
        int4 p0 = *reinterpret_cast<const int4*>(p + 0);
        int4 p1 = *reinterpret_cast<const int4*>(p + 32);
        v8i bv;
        bv[0] = p0.x; bv[1] = p0.y; bv[2] = p0.z; bv[3] = p0.w;
        bv[4] = p1.x; bv[5] = p1.y; bv[6] = p1.z; bv[7] = p1.w;
        bF[sn] = bv;
    }
    #pragma unroll
    for (int sm = 0; sm < 2; ++sm)
        #pragma unroll
        for (int sn = 0; sn < 4; ++sn)
            acc[sm][sn] = __builtin_amdgcn_wmma_i32_16x16x64_iu8(
                 true, aF[sm],  true, bF[sn],
                acc[sm][sn],  false,  false);
}

__global__ __launch_bounds__(256)
void q8_gemm_wmma(const int8_t* __restrict__ x,
                  const int8_t* __restrict__ w,
                  const int32_t* __restrict__ bias,
                  const float* __restrict__ wscale,
                  int8_t* __restrict__ out)
{
    __shared__ int8_t As[2][BM * LDSS];
    __shared__ int8_t Bs[2][BN * LDSS];

    const int tid  = threadIdx.x;
    const int lane = tid & 31;
    const int wave = tid >> 5;
    const int wm   = wave >> 1;
    const int wn   = wave & 1;
    const int lid  = lane & 15;
    const int half = lane >> 4;

    const int mBase = blockIdx.x * BM;
    const int nBase = blockIdx.y * BN;

    const int c0 = tid;
    const int r0 = c0 >> 2, o0 = (c0 & 3) * 16;
    const int c1 = tid + 256;
    const int r1 = c1 >> 2, o1 = (c1 & 3) * 16;

    const int8_t* aG0 = x + (size_t)(mBase + r0) * IN_F + o0;
    const int8_t* aG1 = x + (size_t)(mBase + r1) * IN_F + o1;
    const int8_t* bG0 = w + (size_t)(nBase + r0) * IN_F + o0;
    const int8_t* bG1 = w + (size_t)(nBase + r1) * IN_F + o1;

    const int la0 = r0 * LDSS + o0;
    const int la1 = r1 * LDSS + o1;

    const int aBase = (wm * 32 + lid) * LDSS + half * 16;
    const int bBase = (wn * 64 + lid) * LDSS + half * 16;

    v8i acc[2][4];
    #pragma unroll
    for (int i = 0; i < 2; ++i)
        #pragma unroll
        for (int j = 0; j < 4; ++j) acc[i][j] = 0;

    cp16(aG0, &As[0][la0]);
    cp16(aG1, &As[0][la1]);
    cp16(bG0, &Bs[0][la0]);
    cp16(bG1, &Bs[0][la1]);
    async_wait0();
    __syncthreads();

    #pragma clang loop unroll(disable)
    for (int kt = 0; kt < KITERS - 1; ++kt) {
        const int cur  = kt & 1;
        const int nxt  = cur ^ 1;
        const int koff = (kt + 1) * BK;

        cp16(aG0 + koff, &As[nxt][la0]);
        cp16(aG1 + koff, &As[nxt][la1]);
        cp16(bG0 + koff, &Bs[nxt][la0]);
        cp16(bG1 + koff, &Bs[nxt][la1]);

        frag_compute(As[cur], Bs[cur], aBase, bBase, acc);

        async_wait0();
        __syncthreads();
    }

    frag_compute(As[(KITERS - 1) & 1], Bs[(KITERS - 1) & 1], aBase, bBase, acc);

    #pragma unroll
    for (int sn = 0; sn < 4; ++sn) {
        const int gn = nBase + wn * 64 + sn * 16 + lid;
        const float s = 0.4f * wscale[gn];
        const int   bz = bias[gn];
        #pragma unroll
        for (int sm = 0; sm < 2; ++sm) {
            const int rowBase = mBase + wm * 32 + sm * 16 + half * 8;
            #pragma unroll
            for (int r = 0; r < 8; ++r) {
                float v = (float)(acc[sm][sn][r] + bz) * s;
                v = __builtin_rintf(v);
                v = fminf(fmaxf(v, -128.0f), 127.0f);
                out[(size_t)(rowBase + r) * OUT_F + gn] = (int8_t)(int)v;
            }
        }
    }
}

extern "C" void kernel_launch(void* const* d_in, const int* in_sizes, int n_in,
                              void* d_out, int out_size, void* d_ws, size_t ws_size,
                              hipStream_t stream) {
    (void)in_sizes; (void)n_in; (void)out_size; (void)d_ws; (void)ws_size;
    const int8_t*  x  = (const int8_t*) d_in[0];
    const int8_t*  w  = (const int8_t*) d_in[1];
    const int32_t* b  = (const int32_t*)d_in[2];
    const float*   ws = (const float*)  d_in[3];
    int8_t* out = (int8_t*)d_out;

    dim3 grid(TOKENS / BM, OUT_F / BN);
    dim3 block(256);
    hipLaunchKernelGGL(q8_gemm_wmma, grid, block, 0, stream, x, w, b, ws, out);
}
